// DGATGRU_20572893347935
// MI455X (gfx1250) — matrix operand order repaired and checked by reading, not compiled, not run on hardware
//
#include <hip/hip_runtime.h>
#include <stdint.h>

#define NMSG 16384
#define NNB 10
#define IN_DIM 135
#define HID 256
#define NHEADS 4
#define KPAD_IN 160   // 135 padded to multiple of 32

typedef __attribute__((ext_vector_type(16))) __bf16 v16bf;
typedef __attribute__((ext_vector_type(8)))  __bf16 v8bf;
typedef __attribute__((ext_vector_type(8)))  float  v8f;

#define V8F_ZERO {0.f,0.f,0.f,0.f,0.f,0.f,0.f,0.f}

// ---- WMMA fragment loaders (bf16 16x16x32) -------------------------------
// A-matrix: lane l -> row m = l&15, khalf g = l>>4.
// element e: K = (e>>3)*16 + (e&7) + g*8  => two contiguous 8-elem chunks.
__device__ inline v16bf frag_a(const __bf16* row, int kb, int g) {
  v8bf a0 = *(const v8bf*)(row + kb + g * 8);
  v8bf a1 = *(const v8bf*)(row + kb + 16 + g * 8);
  return __builtin_shufflevector(a0, a1, 0,1,2,3,4,5,6,7,8,9,10,11,12,13,14,15);
}
// B-matrix (K x N from W[n][k] row-major): lane l -> col n = l&15,
// element e: K = (e>>3)*16 + (e&7) + (l>>4)*8  => two contiguous 8-elem chunks.
__device__ inline v16bf frag_b(const __bf16* row, int kb, int g) {
  v8bf b0 = *(const v8bf*)(row + kb + g * 8);
  v8bf b1 = *(const v8bf*)(row + kb + 16 + g * 8);
  return __builtin_shufflevector(b0, b1, 0,1,2,3,4,5,6,7,8,9,10,11,12,13,14,15);
}

struct WPtr { const __bf16 *p0, *p1, *p2, *p3; };
__device__ inline WPtr wptrs(const __bf16* W, int ldw, int tq, int rc) {
  const __bf16* b = W + (size_t)(tq * 64 + rc) * ldw;
  WPtr w; w.p0 = b; w.p1 = b + (size_t)16 * ldw;
  w.p2 = b + (size_t)32 * ldw; w.p3 = b + (size_t)48 * ldw;
  return w;
}
// 4 column tiles (64 cols) accumulated against one shared A fragment
__device__ inline void acc4(v16bf av, const WPtr& w, int kb, int g, v8f acc[4]) {
  acc[0] = __builtin_amdgcn_wmma_f32_16x16x32_bf16(
      false, av, false, frag_b(w.p0, kb, g), (short)0, acc[0], false, false);
  acc[1] = __builtin_amdgcn_wmma_f32_16x16x32_bf16(
      false, av, false, frag_b(w.p1, kb, g), (short)0, acc[1], false, false);
  acc[2] = __builtin_amdgcn_wmma_f32_16x16x32_bf16(
      false, av, false, frag_b(w.p2, kb, g), (short)0, acc[2], false, false);
  acc[3] = __builtin_amdgcn_wmma_f32_16x16x32_bf16(
      false, av, false, frag_b(w.p3, kb, g), (short)0, acc[3], false, false);
}

// ---- GEMM: C = A.W^T + bias (precompute: Fz, Fr, Fh) ----------------------
__global__ void k_gemm_bias(const __bf16* __restrict__ A, int lda,
                            const __bf16* __restrict__ W, int ldw,
                            const float* __restrict__ bias,
                            float* __restrict__ C, int M, int Kpad) {
  int wave = blockIdx.x * (blockDim.x >> 5) + (threadIdx.x >> 5);
  int lane = threadIdx.x & 31;
  int tq = wave & 3, tm = wave >> 2;
  if (tm * 16 >= M) return;
  int g = lane >> 4, rc = lane & 15;
  const __bf16* arow = A + (size_t)(tm * 16 + rc) * lda;
  WPtr w = wptrs(W, ldw, tq, rc);
  v8f acc[4] = {V8F_ZERO, V8F_ZERO, V8F_ZERO, V8F_ZERO};
  for (int kb = 0; kb < Kpad; kb += 32) acc4(frag_a(arow, kb, g), w, kb, g, acc);
#pragma unroll
  for (int s = 0; s < 4; ++s) {
    int n = tq * 64 + s * 16 + rc;
    float b = bias[n];
#pragma unroll
    for (int r = 0; r < 8; ++r)
      C[(size_t)(tm * 16 + r + g * 8) * HID + n] = acc[s][r] + b;
  }
}

// ---- Score GEMM (precompute: sq): one wave covers head tq -----------------
__global__ void k_gemm_score(const __bf16* __restrict__ A, int lda,
                             const __bf16* __restrict__ W, int ldw,
                             const float* __restrict__ bias,
                             const float* __restrict__ alpha, int aoff,
                             float* __restrict__ out, int M, int Kpad) {
  int wave = blockIdx.x * (blockDim.x >> 5) + (threadIdx.x >> 5);
  int lane = threadIdx.x & 31;
  int tq = wave & 3, tm = wave >> 2;
  if (tm * 16 >= M) return;
  int g = lane >> 4, rc = lane & 15;
  const __bf16* arow = A + (size_t)(tm * 16 + rc) * lda;
  WPtr w = wptrs(W, ldw, tq, rc);
  v8f acc[4] = {V8F_ZERO, V8F_ZERO, V8F_ZERO, V8F_ZERO};
  for (int kb = 0; kb < Kpad; kb += 32) acc4(frag_a(arow, kb, g), w, kb, g, acc);
  float bn[4], al[4];
#pragma unroll
  for (int s = 0; s < 4; ++s) {
    int n = tq * 64 + s * 16 + rc;
    bn[s] = bias[n];
    al[s] = alpha[tq * 128 + aoff + s * 16 + rc];
  }
#pragma unroll
  for (int r = 0; r < 8; ++r) {
    float c = 0.f;
#pragma unroll
    for (int s = 0; s < 4; ++s) {
      float v = acc[s][r] + bn[s];
      c += ((v > 0.f) ? v : 0.01f * v) * al[s];
    }
    c += __shfl_xor(c, 1, 32); c += __shfl_xor(c, 2, 32);
    c += __shfl_xor(c, 4, 32); c += __shfl_xor(c, 8, 32);
    if (rc == 0)
      atomicAdd(&out[(size_t)(tm * 16 + r + g * 8) * NHEADS + tq], c);
  }
}

// ---- Fused K-score + V GEMM: shared A = h_bf; V stored bf16 ---------------
__global__ void k_gemm_kv(const __bf16* __restrict__ A,
                          const __bf16* __restrict__ Wk,
                          const __bf16* __restrict__ Wv,
                          const float* __restrict__ bk,
                          const float* __restrict__ bv,
                          const float* __restrict__ alpha,
                          float* __restrict__ sk, __bf16* __restrict__ V) {
  int wave = blockIdx.x * (blockDim.x >> 5) + (threadIdx.x >> 5);
  int lane = threadIdx.x & 31;
  int tq = wave & 3, tm = wave >> 2;
  if (tm * 16 >= NMSG) return;
  int g = lane >> 4, rc = lane & 15;
  const __bf16* arow = A + (size_t)(tm * 16 + rc) * HID;
  WPtr wk = wptrs(Wk, HID, tq, rc);
  WPtr wv = wptrs(Wv, HID, tq, rc);
  v8f ck[4] = {V8F_ZERO, V8F_ZERO, V8F_ZERO, V8F_ZERO};
  v8f cv[4] = {V8F_ZERO, V8F_ZERO, V8F_ZERO, V8F_ZERO};
  for (int kb = 0; kb < HID; kb += 32) {
    v16bf av = frag_a(arow, kb, g);
    acc4(av, wk, kb, g, ck);
    acc4(av, wv, kb, g, cv);
  }
  // V epilogue (bf16 -> halves attention gather traffic)
#pragma unroll
  for (int s = 0; s < 4; ++s) {
    int n = tq * 64 + s * 16 + rc;
    float b = bv[n];
#pragma unroll
    for (int r = 0; r < 8; ++r)
      V[(size_t)(tm * 16 + r + g * 8) * HID + n] = (__bf16)(cv[s][r] + b);
  }
  // score epilogue (head == tq): lrelu . alpha_k, reduce over 64 cols
  float bn[4], al[4];
#pragma unroll
  for (int s = 0; s < 4; ++s) {
    int n = tq * 64 + s * 16 + rc;
    bn[s] = bk[n];
    al[s] = alpha[tq * 128 + 64 + s * 16 + rc];
  }
#pragma unroll
  for (int r = 0; r < 8; ++r) {
    float c = 0.f;
#pragma unroll
    for (int s = 0; s < 4; ++s) {
      float v = ck[s][r] + bn[s];
      c += ((v > 0.f) ? v : 0.01f * v) * al[s];
    }
    c += __shfl_xor(c, 1, 32); c += __shfl_xor(c, 2, 32);
    c += __shfl_xor(c, 4, 32); c += __shfl_xor(c, 8, 32);
    if (rc == 0)
      atomicAdd(&sk[(size_t)(tm * 16 + r + g * 8) * NHEADS + tq], c);
  }
}

// ---- Fused Z + R GEMM + gates: shared A = sumh_bf -------------------------
__global__ void k_gemm_zr(const __bf16* __restrict__ A,
                          const __bf16* __restrict__ Wz2,
                          const __bf16* __restrict__ Ur2,
                          const float* __restrict__ Fz,
                          const float* __restrict__ Fr,
                          const float* __restrict__ sumh,
                          float* __restrict__ zbuf,
                          __bf16* __restrict__ rg) {
  int wave = blockIdx.x * (blockDim.x >> 5) + (threadIdx.x >> 5);
  int lane = threadIdx.x & 31;
  int tq = wave & 3, tm = wave >> 2;
  if (tm * 16 >= NMSG) return;
  int g = lane >> 4, rc = lane & 15;
  const __bf16* arow = A + (size_t)(tm * 16 + rc) * HID;
  WPtr wz = wptrs(Wz2, HID, tq, rc);
  WPtr wu = wptrs(Ur2, HID, tq, rc);
  v8f cz[4] = {V8F_ZERO, V8F_ZERO, V8F_ZERO, V8F_ZERO};
  v8f cr[4] = {V8F_ZERO, V8F_ZERO, V8F_ZERO, V8F_ZERO};
  for (int kb = 0; kb < HID; kb += 32) {
    v16bf av = frag_a(arow, kb, g);
    acc4(av, wz, kb, g, cz);
    acc4(av, wu, kb, g, cr);
  }
#pragma unroll
  for (int s = 0; s < 4; ++s) {
    int n = tq * 64 + s * 16 + rc;
#pragma unroll
    for (int r = 0; r < 8; ++r) {
      size_t i = (size_t)(tm * 16 + r + g * 8) * HID + n;
      float z = 1.f / (1.f + __expf(-(Fz[i] + cz[s][r])));
      float rr = 1.f / (1.f + __expf(-(Fr[i] + cr[s][r])));
      zbuf[i] = z;
      rg[i] = (__bf16)(rr * sumh[i]);
    }
  }
}

// ---- Fused H GEMM + GRU update: A = rg_bf ---------------------------------
__global__ void k_gemm_h(const __bf16* __restrict__ A,
                         const __bf16* __restrict__ Wh2,
                         const float* __restrict__ Fh,
                         const float* __restrict__ zbuf,
                         const float* __restrict__ sumh,
                         float* __restrict__ h, __bf16* __restrict__ hbf,
                         float* __restrict__ rs) {
  int wave = blockIdx.x * (blockDim.x >> 5) + (threadIdx.x >> 5);
  int lane = threadIdx.x & 31;
  int tq = wave & 3, tm = wave >> 2;
  if (tm * 16 >= NMSG) return;
  int g = lane >> 4, rc = lane & 15;
  const __bf16* arow = A + (size_t)(tm * 16 + rc) * HID;
  WPtr w = wptrs(Wh2, HID, tq, rc);
  v8f ch[4] = {V8F_ZERO, V8F_ZERO, V8F_ZERO, V8F_ZERO};
  for (int kb = 0; kb < HID; kb += 32) acc4(frag_a(arow, kb, g), w, kb, g, ch);
#pragma unroll
  for (int r = 0; r < 8; ++r) {
    int m = tm * 16 + r + g * 8;
    float part = 0.f;
#pragma unroll
    for (int s = 0; s < 4; ++s) {
      int n = tq * 64 + s * 16 + rc;
      size_t i = (size_t)m * HID + n;
      float ph = tanhf(Fh[i] + ch[s][r]);
      float z = zbuf[i];
      float sh = sumh[i];
      float nh = (1.f - z) * sh + z * ph;
      if (m == 0) nh = 0.f;              // node_mask zeroes row 0
      h[i] = nh;
      hbf[i] = (__bf16)nh;
      part += nh;
    }
    part += __shfl_xor(part, 1, 32); part += __shfl_xor(part, 2, 32);
    part += __shfl_xor(part, 4, 32); part += __shfl_xor(part, 8, 32);
    if (rc == 0) atomicAdd(&rs[m], part);
  }
}

// ---- Attention: one wave per node; softmax over 10 neighbors per head -----
__global__ void k_attn(const int* __restrict__ bgraph,
                       const float* __restrict__ sq,
                       const float* __restrict__ sk,
                       const float* __restrict__ abias,
                       const float* __restrict__ rs,
                       const __bf16* __restrict__ V,
                       float* __restrict__ sumh,
                       __bf16* __restrict__ sumh_bf) {
  int wave = blockIdx.x * (blockDim.x >> 5) + (threadIdx.x >> 5);
  int lane = threadIdx.x & 31;
  if (wave >= NMSG) return;
  int c0 = lane * 8;
  int hd = c0 >> 6;
  int idx[NNB];
#pragma unroll
  for (int m = 0; m < NNB; ++m) idx[m] = bgraph[wave * NNB + m];
  // prefetch gathered V rows (global_prefetch_b8) to overlap with softmax
#pragma unroll
  for (int m = 0; m < NNB; ++m)
    __builtin_prefetch(V + (size_t)idx[m] * HID + c0, 0, 3);
  float sqv = sq[(size_t)wave * NHEADS + hd];
  float ab = abias[hd];
  float sc[NNB];
  float mx = -3.0e38f;
#pragma unroll
  for (int m = 0; m < NNB; ++m) {
    float s = (rs[idx[m]] == 0.0f)
                  ? -1.0e18f
                  : (sqv + sk[(size_t)idx[m] * NHEADS + hd] + ab);
    sc[m] = s;
    mx = fmaxf(mx, s);
  }
  float se = 0.f;
#pragma unroll
  for (int m = 0; m < NNB; ++m) { sc[m] = __expf(sc[m] - mx); se += sc[m]; }
  float inv = 1.0f / se;
  float acc[8] = {0.f, 0.f, 0.f, 0.f, 0.f, 0.f, 0.f, 0.f};
#pragma unroll
  for (int m = 0; m < NNB; ++m) {
    float w = sc[m] * inv;
    v8bf vv = *(const v8bf*)(V + (size_t)idx[m] * HID + c0);
#pragma unroll
    for (int j = 0; j < 8; ++j) acc[j] += w * (float)vv[j];
  }
  size_t base = (size_t)wave * HID + c0;
#pragma unroll
  for (int j = 0; j < 8; ++j) {
    sumh[base + j] = acc[j];
    sumh_bf[base + j] = (__bf16)acc[j];
  }
}

// ---- f32 -> bf16 convert with column slice + zero pad ---------------------
__global__ void k_cvt(const float* __restrict__ src, int src_ld, int col0,
                      int cols, __bf16* __restrict__ dst, int dst_ld,
                      int total) {
  int i = blockIdx.x * blockDim.x + threadIdx.x;
  if (i >= total) return;
  int r = i / dst_ld, c = i % dst_ld;
  float v = (c < cols) ? src[(size_t)r * src_ld + col0 + c] : 0.f;
  dst[i] = (__bf16)v;
}

__global__ void k_zero(uint32_t* __restrict__ p, size_t n) {
  size_t i = (size_t)blockIdx.x * blockDim.x + threadIdx.x;
  if (i < n) p[i] = 0u;
}

extern "C" void kernel_launch(void* const* d_in, const int* in_sizes, int n_in,
                              void* d_out, int out_size, void* d_ws,
                              size_t ws_size, hipStream_t stream) {
  (void)in_sizes; (void)n_in; (void)out_size; (void)ws_size;
  const float* fmess = (const float*)d_in[0];
  const int*   bgraph= (const int*)d_in[1];
  const float* Wq = (const float*)d_in[2];
  const float* bq = (const float*)d_in[3];
  const float* Wk = (const float*)d_in[4];
  const float* bk = (const float*)d_in[5];
  const float* Wv = (const float*)d_in[6];
  const float* bv = (const float*)d_in[7];
  const float* alpha = (const float*)d_in[8];
  const float* abias = (const float*)d_in[9];
  const float* Wz = (const float*)d_in[10];
  const float* bz = (const float*)d_in[11];
  const float* Wr = (const float*)d_in[12];
  const float* Ur = (const float*)d_in[13];
  const float* bur= (const float*)d_in[14];
  const float* Wh = (const float*)d_in[15];
  const float* bh = (const float*)d_in[16];
  float* h = (float*)d_out;

  char* ws = (char*)d_ws;
  size_t off = 0;
  auto alloc = [&](size_t bytes) -> void* {
    void* p = (void*)(ws + off);
    off = (off + bytes + 255) & ~(size_t)255;
    return p;
  };
  __bf16* h_bf     = (__bf16*)alloc((size_t)NMSG * HID * 2);
  __bf16* fmess_bf = (__bf16*)alloc((size_t)NMSG * KPAD_IN * 2);
  __bf16* sumh_bf  = (__bf16*)alloc((size_t)NMSG * HID * 2);
  __bf16* rg_bf    = (__bf16*)alloc((size_t)NMSG * HID * 2);
  __bf16* Vbf      = (__bf16*)alloc((size_t)NMSG * HID * 2);
  float* sumh = (float*)alloc((size_t)NMSG * HID * 4);
  float* zbuf = (float*)alloc((size_t)NMSG * HID * 4);
  float* Fz   = (float*)alloc((size_t)NMSG * HID * 4);
  float* Fr   = (float*)alloc((size_t)NMSG * HID * 4);
  float* Fh   = (float*)alloc((size_t)NMSG * HID * 4);
  float* sq = (float*)alloc((size_t)NMSG * NHEADS * 4);
  float* sk = (float*)alloc((size_t)NMSG * NHEADS * 4);
  float* rs = (float*)alloc((size_t)NMSG * 4);
  __bf16* Wq_bf  = (__bf16*)alloc(256 * KPAD_IN * 2);
  __bf16* Wk_bf  = (__bf16*)alloc(256 * 256 * 2);
  __bf16* Wv_bf  = (__bf16*)alloc(256 * 256 * 2);
  __bf16* WzA_bf = (__bf16*)alloc(256 * KPAD_IN * 2);
  __bf16* WzB_bf = (__bf16*)alloc(256 * 256 * 2);
  __bf16* Wr_bf  = (__bf16*)alloc(256 * KPAD_IN * 2);
  __bf16* Ur_bf  = (__bf16*)alloc(256 * 256 * 2);
  __bf16* WhA_bf = (__bf16*)alloc(256 * KPAD_IN * 2);
  __bf16* WhB_bf = (__bf16*)alloc(256 * 256 * 2);

  const int T = 256;
  const int GB = (NMSG / 16) * (HID / 64) / 8;   // 4096 waves / 8 per block
  const int WB = NMSG / 8;

  // ---- one-time conversions -------------------------------------------
  { int tot = NMSG * KPAD_IN;
    k_cvt<<<(tot + T - 1) / T, T, 0, stream>>>(fmess, IN_DIM, 0, IN_DIM, fmess_bf, KPAD_IN, tot); }
  { int tot = 256 * KPAD_IN;
    k_cvt<<<(tot + T - 1) / T, T, 0, stream>>>(Wq, IN_DIM, 0, IN_DIM, Wq_bf, KPAD_IN, tot);
    k_cvt<<<(tot + T - 1) / T, T, 0, stream>>>(Wz, IN_DIM + HID, 0, IN_DIM, WzA_bf, KPAD_IN, tot);
    k_cvt<<<(tot + T - 1) / T, T, 0, stream>>>(Wr, IN_DIM, 0, IN_DIM, Wr_bf, KPAD_IN, tot);
    k_cvt<<<(tot + T - 1) / T, T, 0, stream>>>(Wh, IN_DIM + HID, 0, IN_DIM, WhA_bf, KPAD_IN, tot); }
  { int tot = 256 * 256;
    k_cvt<<<(tot + T - 1) / T, T, 0, stream>>>(Wk, HID, 0, HID, Wk_bf, HID, tot);
    k_cvt<<<(tot + T - 1) / T, T, 0, stream>>>(Wv, HID, 0, HID, Wv_bf, HID, tot);
    k_cvt<<<(tot + T - 1) / T, T, 0, stream>>>(Wz, IN_DIM + HID, IN_DIM, HID, WzB_bf, HID, tot);
    k_cvt<<<(tot + T - 1) / T, T, 0, stream>>>(Ur, HID, 0, HID, Ur_bf, HID, tot);
    k_cvt<<<(tot + T - 1) / T, T, 0, stream>>>(Wh, IN_DIM + HID, IN_DIM, HID, WhB_bf, HID, tot); }

  // ---- init state: h=0 (bf16), rowsum=0, sq=0 -------------------------
  k_zero<<<((size_t)NMSG * HID / 2 + T - 1) / T, T, 0, stream>>>((uint32_t*)h_bf, (size_t)NMSG * HID / 2);
  k_zero<<<(NMSG + T - 1) / T, T, 0, stream>>>((uint32_t*)rs, NMSG);
  k_zero<<<(NMSG * NHEADS + T - 1) / T, T, 0, stream>>>((uint32_t*)sq, NMSG * NHEADS);

  // ---- loop-invariant precompute --------------------------------------
  k_gemm_bias<<<GB, T, 0, stream>>>(fmess_bf, KPAD_IN, WzA_bf, KPAD_IN, bz, Fz, NMSG, KPAD_IN);
  k_gemm_bias<<<GB, T, 0, stream>>>(fmess_bf, KPAD_IN, Wr_bf, KPAD_IN, bur, Fr, NMSG, KPAD_IN);
  k_gemm_bias<<<GB, T, 0, stream>>>(fmess_bf, KPAD_IN, WhA_bf, KPAD_IN, bh, Fh, NMSG, KPAD_IN);
  k_gemm_score<<<GB, T, 0, stream>>>(fmess_bf, KPAD_IN, Wq_bf, KPAD_IN, bq, alpha, 0, sq, NMSG, KPAD_IN);

  // ---- DEPTH message-passing steps ------------------------------------
  for (int step = 0; step < 5; ++step) {
    k_zero<<<(NMSG * NHEADS + T - 1) / T, T, 0, stream>>>((uint32_t*)sk, NMSG * NHEADS);
    k_gemm_kv<<<GB, T, 0, stream>>>(h_bf, Wk_bf, Wv_bf, bk, bv, alpha, sk, Vbf);
    k_attn<<<WB, T, 0, stream>>>(bgraph, sq, sk, abias, rs, Vbf, sumh, sumh_bf);
    k_gemm_zr<<<GB, T, 0, stream>>>(sumh_bf, WzB_bf, Ur_bf, Fz, Fr, sumh, zbuf, rg_bf);
    k_zero<<<(NMSG + T - 1) / T, T, 0, stream>>>((uint32_t*)rs, NMSG);
    k_gemm_h<<<GB, T, 0, stream>>>(rg_bf, WhB_bf, Fh, zbuf, sumh, h, h_bf, rs);
  }
}
